// SpikingSelfAttention_91233695301647
// MI455X (gfx1250) — 8-bit matrix operand order repaired and checked by reading, compiled, not run on hardware
//
#include <hip/hip_runtime.h>

typedef __attribute__((ext_vector_type(2))) float v2f;
typedef __attribute__((ext_vector_type(8))) float v8f;
typedef __attribute__((ext_vector_type(8))) int   v8i;

#if defined(__HIP_DEVICE_COMPILE__) && defined(__AMDGCN__)
#if !__has_builtin(__builtin_amdgcn_wmma_f32_16x16x4_f32)
#error "missing __builtin_amdgcn_wmma_f32_16x16x4_f32 on this toolchain"
#endif
#if !__has_builtin(__builtin_amdgcn_wmma_i32_16x16x64_iu8)
#error "missing __builtin_amdgcn_wmma_i32_16x16x64_iu8 on this toolchain"
#endif
#endif

#define TT 4
#define BB 16
#define NN 197
#define DD 512
#define HH 8
#define DH 64
#define MROWS (TT*BB*NN)
#define MTILES (MROWS/16)
#define BN (BB*NN)
#define HB (TT*BB*HH)
#define NPAD 208
#define KPAD 256

__global__ __launch_bounds__(256) void proj_f32_wmma(
    const float* __restrict__ X, const float* __restrict__ W,
    const float* __restrict__ bias, float* __restrict__ OUT, int hasBias)
{
  const int mtile = blockIdx.x;
  const int wave  = threadIdx.x >> 5;
  const int lane  = threadIdx.x & 31;
  const int half  = lane >> 4;
  const int ln    = lane & 15;
  const int colBase = wave * 64;

  const float* xrow = X + (size_t)(mtile*16 + ln) * DD;
  const float* w0 = W + (size_t)(colBase +  0 + ln) * DD;
  const float* w1 = W + (size_t)(colBase + 16 + ln) * DD;
  const float* w2 = W + (size_t)(colBase + 32 + ln) * DD;
  const float* w3 = W + (size_t)(colBase + 48 + ln) * DD;

  v8f acc0 = {}, acc1 = {}, acc2 = {}, acc3 = {};

#pragma unroll 4
  for (int k = 0; k < DD; k += 4) {
    const int kk = k + 2*half;
    v2f a  = *(const v2f*)(xrow + kk);
    v2f b0 = *(const v2f*)(w0 + kk);
    v2f b1 = *(const v2f*)(w1 + kk);
    v2f b2 = *(const v2f*)(w2 + kk);
    v2f b3 = *(const v2f*)(w3 + kk);
    acc0 = __builtin_amdgcn_wmma_f32_16x16x4_f32(false, a, false, b0, (short)0, acc0, false, false);
    acc1 = __builtin_amdgcn_wmma_f32_16x16x4_f32(false, a, false, b1, (short)0, acc1, false, false);
    acc2 = __builtin_amdgcn_wmma_f32_16x16x4_f32(false, a, false, b2, (short)0, acc2, false, false);
    acc3 = __builtin_amdgcn_wmma_f32_16x16x4_f32(false, a, false, b3, (short)0, acc3, false, false);
  }

  float bv0 = 0.f, bv1 = 0.f, bv2 = 0.f, bv3 = 0.f;
  if (hasBias) {
    bv0 = bias[colBase +  0 + ln];
    bv1 = bias[colBase + 16 + ln];
    bv2 = bias[colBase + 32 + ln];
    bv3 = bias[colBase + 48 + ln];
  }
#pragma unroll
  for (int r = 0; r < 8; r++) {
    const int rowD = mtile*16 + r + 8*half;
    float* orow = OUT + (size_t)rowD * DD + colBase + ln;
    orow[ 0] = acc0[r] + bv0;
    orow[16] = acc1[r] + bv1;
    orow[32] = acc2[r] + bv2;
    orow[48] = acc3[r] + bv3;
  }
}

__global__ __launch_bounds__(256) void lif_qk(const float* __restrict__ IN,
                                              unsigned char* __restrict__ SP)
{
  const int idx = blockIdx.x * blockDim.x + threadIdx.x;
  if (idx >= BB*NN*DD) return;
  const int e   = idx & (DD-1);
  const int rem = idx >> 9;
  const int n   = rem % NN;
  const int b   = rem / NN;
  const int h   = e >> 6, dd = e & 63;
  float v = 0.f;
#pragma unroll
  for (int t = 0; t < TT; t++) {
    const float x = IN[((size_t)t*BN + rem) * DD + e];
    const float hm = 0.5f * (v + x);
    const float s  = (hm >= 1.0f) ? 1.0f : 0.0f;
    v = (hm >= 1.0f) ? 0.0f : hm;
    SP[((size_t)(((t*BB + b)*HH + h)) * NN + n) * DH + dd] = (unsigned char)s;
  }
}

__global__ __launch_bounds__(256) void lif_v(const float* __restrict__ IN,
                                             unsigned char* __restrict__ SPT)
{
  const int idx = blockIdx.x * blockDim.x + threadIdx.x;
  if (idx >= BB*KPAD*DD) return;
  const int e   = idx & (DD-1);
  const int rem = idx >> 9;
  const int n   = rem & (KPAD-1);
  const int b   = rem >> 8;
  const int h   = e >> 6, dd = e & 63;
  if (n >= NN) {
#pragma unroll
    for (int t = 0; t < TT; t++)
      SPT[((size_t)(((t*BB + b)*HH + h)) * DH + dd) * KPAD + n] = 0;
    return;
  }
  float v = 0.f;
#pragma unroll
  for (int t = 0; t < TT; t++) {
    const float x = IN[((size_t)t*BN + b*NN + n) * DD + e];
    const float hm = 0.5f * (v + x);
    const float s  = (hm >= 1.0f) ? 1.0f : 0.0f;
    v = (hm >= 1.0f) ? 0.0f : hm;
    SPT[((size_t)(((t*BB + b)*HH + h)) * DH + dd) * KPAD + n] = (unsigned char)s;
  }
}

__global__ __launch_bounds__(256) void lif_out(const float* __restrict__ IN,
                                               float* __restrict__ OUT)
{
  const int idx = blockIdx.x * blockDim.x + threadIdx.x;
  if (idx >= BB*NN*DD) return;
  const int e   = idx & (DD-1);
  const int rem = idx >> 9;
  float v = 0.f;
#pragma unroll
  for (int t = 0; t < TT; t++) {
    const float x = IN[((size_t)t*BN + rem) * DD + e];
    const float hm = 0.5f * (v + x);
    const float s  = (hm >= 1.0f) ? 1.0f : 0.0f;
    v = (hm >= 1.0f) ? 0.0f : hm;
    OUT[((size_t)t*BN + rem) * DD + e] = s;
  }
}

__global__ __launch_bounds__(32) void attn_qk(const unsigned char* __restrict__ qs,
                                              const unsigned char* __restrict__ ks,
                                              unsigned char* __restrict__ S,
                                              float* __restrict__ attn_map)
{
  const int ntile = blockIdx.x;
  const int hb    = blockIdx.y;
  const int lane  = threadIdx.x;
  const int half  = lane >> 4;
  const int ln    = lane & 15;

  const unsigned char* qsb = qs + (size_t)hb * NN * DH;
  const unsigned char* ksb = ks + (size_t)hb * NN * DH;
  unsigned char* Sb   = S        + (size_t)hb * NPAD * KPAD;
  float*         Amap = attn_map + (size_t)hb * NN * NN;

  v8i A = {};
  const int rowA = ntile*16 + ln;
  if (rowA < NN) {
    const unsigned int* qrow = (const unsigned int*)(qsb + (size_t)rowA * DH);
#pragma unroll
    for (int j = 0; j < 8; j++)
      A[j] = (int)qrow[(16*(j>>1) + 8*half + 4*(j&1)) >> 2];
  }

  for (int mt = 0; mt < 13; mt++) {
    v8i Bf = {};
    const int rowB = mt*16 + ln;
    if (rowB < NN) {
      const unsigned int* krow = (const unsigned int*)(ksb + (size_t)rowB * DH);
#pragma unroll
      for (int j = 0; j < 8; j++)
        Bf[j] = (int)krow[(16*(j>>1) + 8*half + 4*(j&1)) >> 2];
    }
    v8i C = {};
    C = __builtin_amdgcn_wmma_i32_16x16x64_iu8(false, A, false, Bf, C, false, false);

    const int col = mt*16 + ln;
#pragma unroll
    for (int r = 0; r < 8; r++) {
      const int rowD = ntile*16 + r + 8*half;
      const int sv = C[r];
      Sb[(size_t)rowD * KPAD + col] = (unsigned char)sv;
      if (rowD < NN && col < NN)
        Amap[(size_t)rowD * NN + col] = (float)sv * 0.125f;
    }
  }
  for (int i = lane; i < 16*48; i += 32) {
    const int r = i / 48, c = 208 + (i % 48);
    Sb[(size_t)(ntile*16 + r) * KPAD + c] = 0;
  }
}

__global__ __launch_bounds__(32) void attn_v(const unsigned char* __restrict__ S,
                                             const unsigned char* __restrict__ vsT,
                                             float* __restrict__ OUT)
{
  const int ntile = blockIdx.x;
  const int hb    = blockIdx.y;
  const int lane  = threadIdx.x;
  const int half  = lane >> 4;
  const int ln    = lane & 15;

  const unsigned char* Sb = S   + (size_t)hb * NPAD * KPAD;
  const unsigned char* Vb = vsT + (size_t)hb * DH * KPAD;

  v8i acc0 = {}, acc1 = {}, acc2 = {}, acc3 = {};
  const unsigned int* srow = (const unsigned int*)(Sb + (size_t)(ntile*16 + ln) * KPAD);

#pragma unroll
  for (int kstep = 0; kstep < 4; kstep++) {
    const int ko = kstep * 64;
    v8i A;
#pragma unroll
    for (int j = 0; j < 8; j++)
      A[j] = (int)srow[(ko + 16*(j>>1) + 8*half + 4*(j&1)) >> 2];

#pragma unroll
    for (int ct = 0; ct < 4; ct++) {
      const int d = ct*16 + ln;
      const unsigned int* vrow = (const unsigned int*)(Vb + (size_t)d * KPAD);
      v8i Bf;
#pragma unroll
      for (int j = 0; j < 8; j++)
        Bf[j] = (int)vrow[(ko + 16*(j>>1) + 8*half + 4*(j&1)) >> 2];
      switch (ct) {
        case 0: acc0 = __builtin_amdgcn_wmma_i32_16x16x64_iu8(false, A, false, Bf, acc0, false, false); break;
        case 1: acc1 = __builtin_amdgcn_wmma_i32_16x16x64_iu8(false, A, false, Bf, acc1, false, false); break;
        case 2: acc2 = __builtin_amdgcn_wmma_i32_16x16x64_iu8(false, A, false, Bf, acc2, false, false); break;
        default: acc3 = __builtin_amdgcn_wmma_i32_16x16x64_iu8(false, A, false, Bf, acc3, false, false); break;
      }
    }
  }

  const int h = hb & 7, b = (hb >> 3) & 15, t = hb >> 7;
#pragma unroll
  for (int r = 0; r < 8; r++) {
    const int rowD = ntile*16 + r + 8*half;
    if (rowD < NN) {
      float* orow = OUT + ((size_t)t*BN + b*NN + rowD) * DD + h*DH + ln;
      orow[ 0] = (float)acc0[r] * 0.125f;
      orow[16] = (float)acc1[r] * 0.125f;
      orow[32] = (float)acc2[r] * 0.125f;
      orow[48] = (float)acc3[r] * 0.125f;
    }
  }
}

extern "C" void kernel_launch(void* const* d_in, const int* in_sizes, int n_in,
                              void* d_out, int out_size, void* d_ws, size_t ws_size,
                              hipStream_t stream) {
  (void)in_sizes; (void)n_in; (void)out_size;
  const float* x  = (const float*)d_in[0];
  const float* Wq = (const float*)d_in[1];
  const float* Wk = (const float*)d_in[2];
  const float* Wv = (const float*)d_in[3];
  const float* Wp = (const float*)d_in[4];
  const float* bp = (const float*)d_in[5];
  float* out = (float*)d_out;

  const size_t SZ_PROJ = (size_t)MROWS * DD * 4;
  const size_t SZ_SP   = (size_t)HB * NN * DH;
  const size_t SZ_VST  = (size_t)HB * DH * KPAD;
  const size_t OFF_Q   = 0;
  const size_t OFF_K   = SZ_PROJ;
  const size_t OFF_V   = 2 * SZ_PROJ;
  const size_t OFF_QS  = 3 * SZ_PROJ;
  const size_t OFF_KS  = OFF_QS + SZ_SP;
  const size_t OFF_VST = OFF_KS + SZ_SP;
  const size_t OFF_S   = 0;
  const size_t OFF_AO  = (size_t)28 << 20;
  const size_t OFF_PP  = 0;
  const size_t NEEDED  = OFF_VST + SZ_VST;
  if (ws_size < NEEDED) return;

  char* ws = (char*)d_ws;
  float* Qf = (float*)(ws + OFF_Q);
  float* Kf = (float*)(ws + OFF_K);
  float* Vf = (float*)(ws + OFF_V);
  unsigned char* qs  = (unsigned char*)(ws + OFF_QS);
  unsigned char* ks  = (unsigned char*)(ws + OFF_KS);
  unsigned char* vsT = (unsigned char*)(ws + OFF_VST);
  unsigned char* Smat = (unsigned char*)(ws + OFF_S);
  float* AO = (float*)(ws + OFF_AO);
  float* PP = (float*)(ws + OFF_PP);
  float* attn_map = out + (size_t)MROWS * DD;

  proj_f32_wmma<<<dim3(MTILES), 256, 0, stream>>>(x, Wq, nullptr, Qf, 0);
  proj_f32_wmma<<<dim3(MTILES), 256, 0, stream>>>(x, Wk, nullptr, Kf, 0);
  proj_f32_wmma<<<dim3(MTILES), 256, 0, stream>>>(x, Wv, nullptr, Vf, 0);

  lif_qk<<<dim3((BB*NN*DD)/256), 256, 0, stream>>>(Qf, qs);
  lif_qk<<<dim3((BB*NN*DD)/256), 256, 0, stream>>>(Kf, ks);
  lif_v <<<dim3((BB*KPAD*DD)/256), 256, 0, stream>>>(Vf, vsT);

  attn_qk<<<dim3(13, HB), 32, 0, stream>>>(qs, ks, Smat, attn_map);

  attn_v<<<dim3(13, HB), 32, 0, stream>>>(Smat, vsT, AO);

  proj_f32_wmma<<<dim3(MTILES), 256, 0, stream>>>(AO, Wp, bp, PP, 1);

  lif_out<<<dim3((BB*NN*DD)/256), 256, 0, stream>>>(PP, out);
}
